// GraphStateEncoder_24240795419595
// MI455X (gfx1250) — matrix operand order repaired and checked by reading, not compiled, not run on hardware
//
#include <hip/hip_runtime.h>

// ---------------------------------------------------------------------------
// GINE encoder on gfx1250: bf16 WMMA (f32 accum) for all large GEMMs.
// Fusion: e_proj_c = ReLU(h1) @ (we2 @ lw_c) + (be2 @ lw_c + lb_c)  => the
// [E,256] edge embedding never touches HBM. Each wave owns M=64 rows
// (4 sub-tiles) so every B fragment is reused by 4 WMMAs (4x less B traffic).
// Single-wave workgroups: 33 KB LDS stage, barriers are no-ops.
// ---------------------------------------------------------------------------

typedef __attribute__((ext_vector_type(16))) __bf16 v16bf;
typedef __attribute__((ext_vector_type(8)))  float  v8f;

union Frag { v16bf v; unsigned u[8]; };
union F8   { v8f v; float f[8]; };

__device__ inline unsigned short bfb(float f) {
  return __builtin_bit_cast(unsigned short, (__bf16)f);   // native cvt, RNE
}
__device__ inline unsigned pack2(float lo, float hi) {
  return (unsigned)bfb(lo) | ((unsigned)bfb(hi) << 16);
}
__device__ inline float relu(float v) { return fmaxf(v, 0.f); }

// 16-bit A-matrix 16x32 layout: lane half selects K block {0..7|8..15},
// VGPR p<4 -> K = hi + 2p, p>=4 -> K = 16 + hi + 2(p-4)  (pairs contiguous).
__device__ inline int kpairA(int p, int lane) {
  int hi = (lane >> 4) << 3;
  return (p < 4) ? (hi + 2 * p) : (8 + hi + 2 * p);
}
// 16-bit B-matrix 32x16 layout: same K order as A (lane half selects K block).
__device__ inline int kpairB(int p, int lane) {
  int hi = (lane >> 4) << 3;
  return (p < 4) ? (hi + 2 * p) : (8 + hi + 2 * p);
}
__device__ inline v8f wmma_bf16(v16bf a, v16bf b, v8f c) {
  return __builtin_amdgcn_wmma_f32_16x16x32_bf16(false, a, false, b, (short)0, c,
                                                 false, false);
}
__device__ inline v8f zero8() { F8 z;
#pragma unroll
  for (int i = 0; i < 8; ++i) z.f[i] = 0.f;
  return z.v; }

// -------------------------- prep kernels -----------------------------------
// we1 padded+transposed: wT[n*32+k] = k<16 ? bf16(we1[k,n]) : 0
__global__ void k_prep_we1(const float* __restrict__ we1,
                           unsigned short* __restrict__ wT) {
  int idx = blockIdx.x * 256 + threadIdx.x;
  if (idx >= 256 * 32) return;
  int n = idx >> 5, k = idx & 31;
  wT[idx] = (k < 16) ? bfb(we1[k * 256 + n]) : (unsigned short)0;
}
// wT[n*K + k] = bf16(w[k*Nc + n])   (transpose so K pairs are contiguous)
__global__ void k_transpose_bf(const float* __restrict__ w,
                               unsigned short* __restrict__ wT, int K, int Nc) {
  int idx = blockIdx.x * 256 + threadIdx.x;
  if (idx >= K * Nc) return;
  int n = idx / K, k = idx % K;
  wT[n * K + k] = bfb(w[k * Nc + n]);
}
// WcT[n*256 + j] = sum_k we2[j,k] * lw[k,n]   (combined weight, transposed)
__global__ void k_combine(const float* __restrict__ we2,
                          const float* __restrict__ lw,
                          unsigned short* __restrict__ WcT, int NC) {
  int idx = blockIdx.x * 256 + threadIdx.x;
  if (idx >= NC * 256) return;
  int n = idx / 256, j = idx % 256;
  float s = 0.f;
  for (int k = 0; k < 256; ++k) s += we2[j * 256 + k] * lw[k * NC + n];
  WcT[n * 256 + j] = bfb(s);
}
__global__ void k_bias(const float* __restrict__ be2, const float* __restrict__ lw,
                       const float* __restrict__ lb, float* __restrict__ bc, int NC) {
  int n = blockIdx.x * 256 + threadIdx.x;
  if (n >= NC) return;
  float s = lb[n];
  for (int k = 0; k < 256; ++k) s += be2[k] * lw[k * NC + n];
  bc[n] = s;
}

// -------------------------- edge pass --------------------------------------
// Per wave: 64 edges (4 sub-tiles). h1 = relu(edge_attr@we1 + be1) -> LDS,
// e_proj = h1 @ WcT + bc; msg = relu(x[src] + e_proj); atomic agg[dst] +=
template <int NC>
__global__ __launch_bounds__(32) void k_edge(
    const float* __restrict__ ea,            // [E,16]
    const int* __restrict__ eidx,            // [2,E]
    const float* __restrict__ xin,           // [N,NC]
    const unsigned short* __restrict__ we1T, // [256][32] bf16, K padded
    const float* __restrict__ be1,           // [256]
    const unsigned short* __restrict__ WcT,  // [NC][256] bf16
    const float* __restrict__ bc,            // [NC]
    float* __restrict__ agg,                 // [N,NC]
    int E, int ntiles16) {
  __shared__ unsigned short stage[64][264];  // 33792 B
  const int lane = threadIdx.x;
  const int base16 = blockIdx.x * 4;
  const int col = lane & 15;
  const int mhi = (lane >> 4) << 3;

  bool act[4];
  int srcOff[4][8], dstOff[4][8];
  Frag a0[4];
#pragma unroll
  for (int s = 0; s < 4; ++s) {
    act[s] = (base16 + s) < ntiles16;
    if (act[s]) {
      const int e0 = (base16 + s) * 16;
#pragma unroll
      for (int r = 0; r < 8; ++r) {          // C/D rows owned by this lane
        int m = r + mhi;
        srcOff[s][r] = eidx[e0 + m] * NC;
        dstOff[s][r] = eidx[E + e0 + m] * NC;
      }
#pragma unroll
      for (int p = 0; p < 8; ++p) {          // edge_attr tile, K padded 16->32
        if (p < 4) {
          int kb = kpairA(p, lane);          // < 16
          a0[s].u[p] = pack2(ea[(e0 + col) * 16 + kb],
                             ea[(e0 + col) * 16 + kb + 1]);
        } else a0[s].u[p] = 0u;
      }
    } else {
#pragma unroll
      for (int p = 0; p < 8; ++p) a0[s].u[p] = 0u;
#pragma unroll
      for (int r = 0; r < 8; ++r) srcOff[s][r] = dstOff[s][r] = 0;
    }
  }
  // GEMM1: h1 = relu(A0 @ we1 + be1), staged to LDS as bf16
  for (int nt = 0; nt < 16; ++nt) {
    int n = col + nt * 16;
    Frag b;
#pragma unroll
    for (int p = 0; p < 8; ++p)
      b.u[p] = *(const unsigned*)&we1T[n * 32 + kpairB(p, lane)];
    float bias = be1[n];
#pragma unroll
    for (int s = 0; s < 4; ++s) {
      if (!act[s]) continue;                 // wave-uniform branch
      F8 c; c.v = wmma_bf16(a0[s].v, b.v, zero8());
#pragma unroll
      for (int r = 0; r < 8; ++r)
        stage[s * 16 + r + mhi][n] = bfb(relu(c.f[r] + bias));
    }
  }
  __syncthreads();                           // no-op (1 wave); ordering fence
  // A fragments for GEMM2 (K = 256), register resident for B reuse
  Frag a2[4][8];
#pragma unroll
  for (int s = 0; s < 4; ++s)
#pragma unroll
    for (int ck = 0; ck < 8; ++ck)
#pragma unroll
      for (int p = 0; p < 8; ++p)
        a2[s][ck].u[p] =
            *(const unsigned*)&stage[s * 16 + col][ck * 32 + kpairA(p, lane)];
  const int NT = NC / 16;
  for (int nt = 0; nt < NT; ++nt) {
    int n = col + nt * 16;
    F8 acc[4];
#pragma unroll
    for (int s = 0; s < 4; ++s) acc[s].v = zero8();
#pragma unroll
    for (int ck = 0; ck < 8; ++ck) {
      Frag b;
#pragma unroll
      for (int p = 0; p < 8; ++p)
        b.u[p] = *(const unsigned*)&WcT[n * 256 + ck * 32 + kpairB(p, lane)];
#pragma unroll
      for (int s = 0; s < 4; ++s)
        acc[s].v = wmma_bf16(a2[s][ck].v, b.v, acc[s].v);
    }
    float bias = bc[n];
#pragma unroll
    for (int s = 0; s < 4; ++s) {
      if (!act[s]) continue;
#pragma unroll
      for (int r = 0; r < 8; ++r) {
        float msg = relu(acc[s].f[r] + bias + xin[srcOff[s][r] + n]);
        atomicAdd(&agg[dstOff[s][r] + n], msg);
      }
    }
  }
}

// -------------------------- node MLP ---------------------------------------
// xout = relu( relu((xin+agg)@w1 + b1) @ w2 + b2 ), 64 nodes per wave
template <int IND>
__global__ __launch_bounds__(32) void k_node(
    const float* __restrict__ xin, const float* __restrict__ agg,
    const unsigned short* __restrict__ w1T,  // [256][IND] bf16
    const float* __restrict__ b1,
    const unsigned short* __restrict__ w2T,  // [256][256] bf16
    const float* __restrict__ b2,
    float* __restrict__ xout,                // [N,256]
    int ntiles16) {
  __shared__ unsigned short stage[64][264];
  const int lane = threadIdx.x;
  const int base16 = blockIdx.x * 4;
  const int col = lane & 15;
  const int mhi = (lane >> 4) << 3;
  const int CH = IND / 32;

  bool act[4];
  Frag a1[4][IND / 32];
#pragma unroll
  for (int s = 0; s < 4; ++s) {
    act[s] = (base16 + s) < ntiles16;
    if (act[s]) {
      int row = (base16 + s) * 16 + col;
#pragma unroll
      for (int ck = 0; ck < CH; ++ck)
#pragma unroll
        for (int p = 0; p < 8; ++p) {
          int k = ck * 32 + kpairA(p, lane);
          a1[s][ck].u[p] = pack2(xin[row * IND + k] + agg[row * IND + k],
                                 xin[row * IND + k + 1] + agg[row * IND + k + 1]);
        }
    } else {
#pragma unroll
      for (int ck = 0; ck < CH; ++ck)
#pragma unroll
        for (int p = 0; p < 8; ++p) a1[s][ck].u[p] = 0u;
    }
  }
  for (int nt = 0; nt < 16; ++nt) {
    int n = col + nt * 16;
    F8 acc[4];
#pragma unroll
    for (int s = 0; s < 4; ++s) acc[s].v = zero8();
#pragma unroll
    for (int ck = 0; ck < CH; ++ck) {
      Frag b;
#pragma unroll
      for (int p = 0; p < 8; ++p)
        b.u[p] = *(const unsigned*)&w1T[n * IND + ck * 32 + kpairB(p, lane)];
#pragma unroll
      for (int s = 0; s < 4; ++s)
        acc[s].v = wmma_bf16(a1[s][ck].v, b.v, acc[s].v);
    }
    float bias = b1[n];
#pragma unroll
    for (int s = 0; s < 4; ++s) {
      if (!act[s]) continue;
#pragma unroll
      for (int r = 0; r < 8; ++r)
        stage[s * 16 + r + mhi][n] = bfb(relu(acc[s].f[r] + bias));
    }
  }
  __syncthreads();
  Frag a2[4][8];
#pragma unroll
  for (int s = 0; s < 4; ++s)
#pragma unroll
    for (int ck = 0; ck < 8; ++ck)
#pragma unroll
      for (int p = 0; p < 8; ++p)
        a2[s][ck].u[p] =
            *(const unsigned*)&stage[s * 16 + col][ck * 32 + kpairA(p, lane)];
  for (int nt = 0; nt < 16; ++nt) {
    int n = col + nt * 16;
    F8 acc[4];
#pragma unroll
    for (int s = 0; s < 4; ++s) acc[s].v = zero8();
#pragma unroll
    for (int ck = 0; ck < 8; ++ck) {
      Frag b;
#pragma unroll
      for (int p = 0; p < 8; ++p)
        b.u[p] = *(const unsigned*)&w2T[n * 256 + ck * 32 + kpairB(p, lane)];
#pragma unroll
      for (int s = 0; s < 4; ++s)
        acc[s].v = wmma_bf16(a2[s][ck].v, b.v, acc[s].v);
    }
    float bias = b2[n];
#pragma unroll
    for (int s = 0; s < 4; ++s) {
      if (!act[s]) continue;
      int n0 = (base16 + s) * 16;
#pragma unroll
      for (int r = 0; r < 8; ++r)
        xout[(n0 + r + mhi) * 256 + n] = relu(acc[s].f[r] + bias);
    }
  }
}

// -------------------------- pooling ----------------------------------------
__global__ void k_pool(const float* __restrict__ x2, const int* __restrict__ batch,
                       float* __restrict__ sums, float* __restrict__ cnt) {
  int node = blockIdx.x, c = threadIdx.x;
  int g = batch[node];
  atomicAdd(&sums[g * 256 + c], x2[node * 256 + c]);
  if (c == 0) atomicAdd(&cnt[g], 1.0f);
}
__global__ void k_final(const float* __restrict__ sums, const float* __restrict__ cnt,
                        float* __restrict__ out) {
  int g = blockIdx.x, c = threadIdx.x;
  float n = cnt[g];
  out[g * 256 + c] = sums[g * 256 + c] / (n > 1.f ? n : 1.f);
}

// -------------------------- launcher ---------------------------------------
extern "C" void kernel_launch(void* const* d_in, const int* in_sizes, int n_in,
                              void* d_out, int out_size, void* d_ws, size_t ws_size,
                              hipStream_t stream) {
  (void)n_in; (void)ws_size;
  const float* x     = (const float*)d_in[0];
  const float* ea    = (const float*)d_in[1];
  const int*   eidx  = (const int*)d_in[2];
  const int*   batch = (const int*)d_in[3];
  const float* we1   = (const float*)d_in[4];
  const float* be1   = (const float*)d_in[5];
  const float* we2   = (const float*)d_in[6];
  const float* be2   = (const float*)d_in[7];
  const float* l0_lw = (const float*)d_in[8];
  const float* l0_lb = (const float*)d_in[9];
  const float* l0_w1 = (const float*)d_in[10];
  const float* l0_b1 = (const float*)d_in[11];
  const float* l0_w2 = (const float*)d_in[12];
  const float* l0_b2 = (const float*)d_in[13];
  const float* l1_lw = (const float*)d_in[14];
  const float* l1_lb = (const float*)d_in[15];
  const float* l1_w1 = (const float*)d_in[16];
  const float* l1_b1 = (const float*)d_in[17];
  const float* l1_w2 = (const float*)d_in[18];
  const float* l1_b2 = (const float*)d_in[19];
  float* out = (float*)d_out;

  const int N = in_sizes[0] / 64;
  const int E = in_sizes[1] / 16;
  const int G = out_size / 256;

  char* ws = (char*)d_ws;
  size_t off = 0;
  auto take = [&](size_t b) { char* p = ws + off;
                              off = (off + b + 255) & ~(size_t)255; return p; };
  unsigned short* we1T = (unsigned short*)take(256 * 32 * 2);
  unsigned short* w01T = (unsigned short*)take(256 * 64 * 2);
  unsigned short* w02T = (unsigned short*)take(256 * 256 * 2);
  unsigned short* w11T = (unsigned short*)take(256 * 256 * 2);
  unsigned short* w12T = (unsigned short*)take(256 * 256 * 2);
  unsigned short* W0cT = (unsigned short*)take(64 * 256 * 2);
  unsigned short* W1cT = (unsigned short*)take(256 * 256 * 2);
  float* b0c  = (float*)take(64 * 4);
  float* b1c  = (float*)take(256 * 4);
  float* agg0 = (float*)take((size_t)N * 64 * 4);
  float* x1   = (float*)take((size_t)N * 256 * 4);
  float* agg1 = (float*)take((size_t)N * 256 * 4);
  float* x2   = (float*)take((size_t)N * 256 * 4);
  float* sums = (float*)take((size_t)G * 256 * 4);
  float* cnt  = (float*)take((size_t)G * 4);

  hipMemsetAsync(agg0, 0, (size_t)N * 64 * 4, stream);
  hipMemsetAsync(agg1, 0, (size_t)N * 256 * 4, stream);
  hipMemsetAsync(sums, 0, (size_t)G * 256 * 4, stream);
  hipMemsetAsync(cnt, 0, (size_t)G * 4, stream);

  k_prep_we1<<<(256 * 32 + 255) / 256, 256, 0, stream>>>(we1, we1T);
  k_transpose_bf<<<(64 * 256 + 255) / 256, 256, 0, stream>>>(l0_w1, w01T, 64, 256);
  k_transpose_bf<<<(256 * 256 + 255) / 256, 256, 0, stream>>>(l0_w2, w02T, 256, 256);
  k_transpose_bf<<<(256 * 256 + 255) / 256, 256, 0, stream>>>(l1_w1, w11T, 256, 256);
  k_transpose_bf<<<(256 * 256 + 255) / 256, 256, 0, stream>>>(l1_w2, w12T, 256, 256);
  k_combine<<<(64 * 256 + 255) / 256, 256, 0, stream>>>(we2, l0_lw, W0cT, 64);
  k_combine<<<(256 * 256 + 255) / 256, 256, 0, stream>>>(we2, l1_lw, W1cT, 256);
  k_bias<<<1, 256, 0, stream>>>(be2, l0_lw, l0_lb, b0c, 64);
  k_bias<<<1, 256, 0, stream>>>(be2, l1_lw, l1_lb, b1c, 256);

  const int etiles16 = E / 16;                     // E = 320000 -> exact
  const int eblocks = (etiles16 + 3) / 4;          // 64 edges per wave/block
  const int ntiles16 = N / 16;                     // N = 10000 -> exact
  const int nblocks = (ntiles16 + 3) / 4;

  k_edge<64><<<eblocks, 32, 0, stream>>>(ea, eidx, x, we1T, be1, W0cT, b0c,
                                         agg0, E, etiles16);
  k_node<64><<<nblocks, 32, 0, stream>>>(x, agg0, w01T, l0_b1, w02T, l0_b2,
                                         x1, ntiles16);
  k_edge<256><<<eblocks, 32, 0, stream>>>(ea, eidx, x1, we1T, be1, W1cT, b1c,
                                          agg1, E, etiles16);
  k_node<256><<<nblocks, 32, 0, stream>>>(x1, agg1, w11T, l1_b1, w12T, l1_b2,
                                          x2, ntiles16);
  k_pool<<<N, 256, 0, stream>>>(x2, batch, sums, cnt);
  k_final<<<G, 256, 0, stream>>>(sums, cnt, out);
}
